// AttentionHead_39556648796488
// MI455X (gfx1250) — matrix operand order repaired and checked by reading, not compiled, not run on hardware
//
#include <hip/hip_runtime.h>

// CDNA5 / gfx1250 flash-attention head.
//   proj: X[f32] @ W[f32] -> f16 Q/K (row-major) and V (transposed) via WMMA.
//   attn: causal flash attention, online softmax, WMMA QK^T + PV.
//         K/V tiles double-buffered in LDS, filled by the Tensor Data Mover
//         (tensor_load_to_lds + s_wait_tensorcnt) with DMA/compute overlap.
// All per-lane WMMA fragments are contiguous 16/32-byte runs -> b128 loads.

typedef __attribute__((ext_vector_type(16))) _Float16 v16h;
typedef __attribute__((ext_vector_type(8)))  _Float16 v8h;
typedef __attribute__((ext_vector_type(8)))  float    v8f;
typedef __attribute__((ext_vector_type(4)))  float    v4f;
typedef __attribute__((ext_vector_type(4)))  unsigned int u32x4;
typedef __attribute__((ext_vector_type(8)))  int      i32x8;
typedef __attribute__((ext_vector_type(4)))  int      i32x4;

#define D_IN  512
#define D_OUT 64
#define SEQ   4096
#define BATCH 4

static __device__ __forceinline__ v8f wmma_f16(v16h a, v16h b, v8f c) {
  return __builtin_amdgcn_wmma_f32_16x16x32_f16(false, a, false, b, (short)0, c,
                                                false, false);
}

static __device__ __forceinline__ v16h cat8(v8h lo, v8h hi) {
  return __builtin_shufflevector(lo, hi, 0, 1, 2, 3, 4, 5, 6, 7, 8, 9, 10, 11,
                                 12, 13, 14, 15);
}

// ---------------------------------------------------------------------------
// Tensor Data Mover: 2-D f16 tile (64 wide x 64 rows) from global -> LDS.
// D# per CDNA5 ISA: group0 = {count=1, lds_addr,
// global_addr, type=2}; group1 = {data_size=2B, tensor dims, tile 64x64,
// tensor_dim0_stride}. Groups 2/3 zero (2-D tensor).
// ---------------------------------------------------------------------------
#if __has_builtin(__builtin_amdgcn_tensor_load_to_lds)
#define USE_TDM 1
static __device__ __forceinline__ void tdm_load_2d_f16(
    unsigned lds_off, const void* gptr, unsigned tensor_d0, unsigned tensor_d1,
    unsigned stride_d0) {
  unsigned long long ga = (unsigned long long)(size_t)gptr;
  u32x4 g0;
  g0[0] = 1u;                                    // count=1 (valid descriptor)
  g0[1] = lds_off;                               // LDS byte address
  g0[2] = (unsigned)ga;                          // global addr [31:0]
  g0[3] = (unsigned)(ga >> 32) | (2u << 30);     // addr [56:32] | type=2

  unsigned long long q0 =
      (1ull << 16)                                   // data_size = 1 -> 2 bytes
      | ((unsigned long long)(tensor_d0 & 0xFFFFu) << 48);  // dim0 low16
  unsigned long long q1 =
      (unsigned long long)(tensor_d0 >> 16)            // dim0 high16
      | ((unsigned long long)tensor_d1 << 16)          // tensor_dim1
      | (64ull << 48);                                 // tile_dim0 = 64
  unsigned long long q2 =
      64ull                                            // tile_dim1 = 64
      | ((unsigned long long)stride_d0 << 32);         // dim0_stride low32
  unsigned long long q3 = 0ull;

  i32x8 g1;
  g1[0] = (int)(unsigned)q0; g1[1] = (int)(q0 >> 32);
  g1[2] = (int)(unsigned)q1; g1[3] = (int)(q1 >> 32);
  g1[4] = (int)(unsigned)q2; g1[5] = (int)(q2 >> 32);
  g1[6] = (int)(unsigned)q3; g1[7] = (int)(q3 >> 32);

  i32x4 z4 = {0, 0, 0, 0};
#if __clang_major__ >= 23
  i32x8 z8 = {0, 0, 0, 0, 0, 0, 0, 0};
  __builtin_amdgcn_tensor_load_to_lds(g0, g1, z4, z4, z8, 0);
#else
  __builtin_amdgcn_tensor_load_to_lds(g0, g1, z4, z4, 0);
#endif
}
#else
#define USE_TDM 0
#endif

// ---------------------------------------------------------------------------
// Projection: Y = X[rows,512] @ W[512,64] -> f16.
// transposed==0: Y[row][e] row-major (Q, K).
// transposed==1: Y[e][row] per batch slice (V): key-contiguous columns.
// 256 threads = 8 waves, wave owns a 16x64 tile. W staged column-major in LDS.
// X is streamed once -> nontemporal loads.
// ---------------------------------------------------------------------------
__global__ __launch_bounds__(256)
void proj_f16_kernel(const float* __restrict__ X, const float* __restrict__ W,
                     _Float16* __restrict__ Y, int transposed) {
  __shared__ __align__(32) _Float16 sWt[D_OUT * D_IN];  // [n][k], 64 KB

  const int tid = threadIdx.x;
  for (int i = tid; i < D_IN * D_OUT; i += 256) {
    const int k = i >> 6, n = i & 63;          // W row-major [k][n]
    sWt[(size_t)n * D_IN + k] = (_Float16)W[i];
  }
  __syncthreads();

  const int lane = tid & 31;
  const int wave = tid >> 5;
  const int rowBase = blockIdx.x * 128 + wave * 16;

  const int m  = lane & 15;
  const int hl = lane >> 4;
  const int kb = hl * 16;

  v8f acc[4] = {};

  for (int kc = 0; kc < D_IN; kc += 32) {
    const float* xr = X + (size_t)(rowBase + m) * D_IN + kc;
    const v4f xa = __builtin_nontemporal_load((const v4f*)(xr + kb));
    const v4f xb = __builtin_nontemporal_load((const v4f*)(xr + kb + 4));
    const v4f xc = __builtin_nontemporal_load((const v4f*)(xr + kb + 8));
    const v4f xd = __builtin_nontemporal_load((const v4f*)(xr + kb + 12));
    v16h a;
    a[0] = (_Float16)xa.x;  a[1] = (_Float16)xa.y;
    a[2] = (_Float16)xa.z;  a[3] = (_Float16)xa.w;
    a[4] = (_Float16)xb.x;  a[5] = (_Float16)xb.y;
    a[6] = (_Float16)xb.z;  a[7] = (_Float16)xb.w;
    a[8]  = (_Float16)xc.x; a[9]  = (_Float16)xc.y;
    a[10] = (_Float16)xc.z; a[11] = (_Float16)xc.w;
    a[12] = (_Float16)xd.x; a[13] = (_Float16)xd.y;
    a[14] = (_Float16)xd.z; a[15] = (_Float16)xd.w;

#pragma unroll
    for (int t = 0; t < 4; ++t) {
      const _Float16* wp = sWt + (size_t)(t * 16 + m) * D_IN + kc + hl * 16;
      const v16h bw = *(const v16h*)wp;
      acc[t] = wmma_f16(a, bw, acc[t]);
    }
  }

  if (!transposed) {
    const int mo = hl * 8;
#pragma unroll
    for (int t = 0; t < 4; ++t)
#pragma unroll
      for (int r = 0; r < 8; ++r)
        Y[(size_t)(rowBase + mo + r) * D_OUT + t * 16 + m] = (_Float16)acc[t][r];
  } else {
    // Vt[b*64 + e][key]; per lane: 8 consecutive keys -> one 16B store.
    const int b = rowBase / SEQ;
    const int keyBase = (rowBase - b * SEQ) + hl * 8;
#pragma unroll
    for (int t = 0; t < 4; ++t) {
      v8h h8;
#pragma unroll
      for (int r = 0; r < 8; ++r) h8[r] = (_Float16)acc[t][r];
      *(v8h*)(Y + ((size_t)b * D_OUT + t * 16 + m) * SEQ + keyBase) = h8;
    }
  }
}

// ---------------------------------------------------------------------------
// Flash attention (causal). Block = 256 threads = 8 waves, 128 Q rows/block.
// K/V tiles double-buffered; wave 0 drives the TDM one tile ahead.
// sK row-major [key][e]; sVt transposed [e][key] (PV B-frags contiguous).
// ---------------------------------------------------------------------------
__global__ __launch_bounds__(256)
void flash_attn_kernel(const _Float16* __restrict__ Qh,
                       const _Float16* __restrict__ Kh,
                       const _Float16* __restrict__ Vt,
                       float* __restrict__ Out) {
  __shared__ __align__(32) _Float16 sK[2][64 * 64];    // 16 KB
  __shared__ __align__(32) _Float16 sVt[2][64 * 64];   // 16 KB
  __shared__ __align__(32) _Float16 sP[8][16 * 64];    // per-wave P, 16 KB

  const int tid  = threadIdx.x;
  const int lane = tid & 31;
  const int wave = tid >> 5;
  const int bIdx = blockIdx.y;
  const int qBlock = blockIdx.x;                  // 128-row q block
  const int qRowBase = qBlock * 128 + wave * 16;  // this wave's 16 q rows

  const int m  = lane & 15;
  const int hl = lane >> 4;
  const int kb = hl * 16;
  const int mo = hl * 8;
  const float scale = 0.015625f;  // 1/sqrt(4096)

  // Q fragments (A layout): two 16B runs per e-chunk.
  v16h qf[2];
  {
    const _Float16* qr0 = Qh + ((size_t)bIdx * SEQ + qRowBase + m) * D_OUT;
#pragma unroll
    for (int c = 0; c < 2; ++c) {
      const v8h lo = *(const v8h*)(qr0 + c * 32 + kb);
      const v8h hi = *(const v8h*)(qr0 + c * 32 + kb + 8);
      qf[c] = cat8(lo, hi);
    }
  }

  v8f o[4] = {};
  float mrow[8], lrow[8];
#pragma unroll
  for (int r = 0; r < 8; ++r) { mrow[r] = -3.0e38f; lrow[r] = 0.0f; }

  const int nTiles = 2 * qBlock + 2;  // causal: tiles covering keys <= q_max

#if USE_TDM
  if (tid < 32) {  // wave 0 drives the DMA; TDM ignores EXEC, SGPR operands
    tdm_load_2d_f16((unsigned)(size_t)&sK[0][0],
                    Kh + (size_t)bIdx * SEQ * D_OUT, D_OUT, SEQ, D_OUT);
    tdm_load_2d_f16((unsigned)(size_t)&sVt[0][0],
                    Vt + (size_t)bIdx * D_OUT * SEQ, SEQ, D_OUT, SEQ);
  }
#endif

  for (int j = 0; j < nTiles; ++j) {
    const int cur = j & 1;
#if USE_TDM
    if (tid < 32) {
      if (j + 1 < nTiles) {
        // Issue next tile into the other buffer, then wait for the current
        // tile's two DMAs (in-order completion => TENSORcnt <= 2).
        tdm_load_2d_f16((unsigned)(size_t)&sK[cur ^ 1][0],
                        Kh + ((size_t)bIdx * SEQ + (j + 1) * 64) * D_OUT,
                        D_OUT, SEQ, D_OUT);
        tdm_load_2d_f16((unsigned)(size_t)&sVt[cur ^ 1][0],
                        Vt + (size_t)bIdx * D_OUT * SEQ + (j + 1) * 64,
                        SEQ, D_OUT, SEQ);
        __builtin_amdgcn_s_wait_tensorcnt(2);
      } else {
        __builtin_amdgcn_s_wait_tensorcnt(0);
      }
    }
#else
    {
      const float4* srcK =
          (const float4*)(Kh + ((size_t)bIdx * SEQ + j * 64) * D_OUT);
      float4* dK = (float4*)&sK[cur][0];
      for (int i = tid; i < (64 * 64) / 8; i += 256) dK[i] = srcK[i];
      const _Float16* vb = Vt + (size_t)bIdx * D_OUT * SEQ + j * 64;
      for (int i = tid; i < (64 * 64) / 8; i += 256) {
        const int e = i >> 3, c4 = i & 7;
        ((float4*)&sVt[cur][0])[i] = *(const float4*)(vb + (size_t)e * SEQ + c4 * 8);
      }
    }
#endif
    __syncthreads();  // tile j visible to all waves

    // S = Q K^T : 4 key-subtiles x 2 e-chunks of WMMA.
    v8f s[4] = {};
#pragma unroll
    for (int t = 0; t < 4; ++t) {
#pragma unroll
      for (int c = 0; c < 2; ++c) {
        const v16h bk = *(const v16h*)(&sK[cur][0] +
                                       (size_t)(t * 16 + m) * D_OUT + c * 32 +
                                       hl * 16);
        s[t] = wmma_f16(qf[c], bk, s[t]);
      }
    }

    // Scale + causal mask.
    const int keyBase = j * 64;
#pragma unroll
    for (int t = 0; t < 4; ++t) {
      const int key = keyBase + t * 16 + m;
#pragma unroll
      for (int r = 0; r < 8; ++r) {
        const int q = qRowBase + mo + r;
        const float sv = s[t][r] * scale;
        s[t][r] = (key <= q) ? sv : -3.0e38f;
      }
    }

    // Online softmax; row r spans 16 lanes of each half -> xor 1,2,4,8.
#pragma unroll
    for (int r = 0; r < 8; ++r) {
      float mx = fmaxf(fmaxf(s[0][r], s[1][r]), fmaxf(s[2][r], s[3][r]));
      mx = fmaxf(mx, __shfl_xor(mx, 1, 32));
      mx = fmaxf(mx, __shfl_xor(mx, 2, 32));
      mx = fmaxf(mx, __shfl_xor(mx, 4, 32));
      mx = fmaxf(mx, __shfl_xor(mx, 8, 32));

      const float mn = fmaxf(mrow[r], mx);
      const float alpha = __expf(mrow[r] - mn);
      mrow[r] = mn;

      float sum = 0.0f;
#pragma unroll
      for (int t = 0; t < 4; ++t) {
        const float p = __expf(s[t][r] - mn);
        s[t][r] = p;
        sum += p;
      }
      sum += __shfl_xor(sum, 1, 32);
      sum += __shfl_xor(sum, 2, 32);
      sum += __shfl_xor(sum, 4, 32);
      sum += __shfl_xor(sum, 8, 32);
      lrow[r] = lrow[r] * alpha + sum;
#pragma unroll
      for (int t = 0; t < 4; ++t) o[t][r] *= alpha;
    }

    // P: C-layout -> row-major in per-wave LDS, reload as A fragments.
#pragma unroll
    for (int t = 0; t < 4; ++t)
#pragma unroll
      for (int r = 0; r < 8; ++r)
        sP[wave][(mo + r) * 64 + t * 16 + m] = (_Float16)s[t][r];

    v16h pf[2];
#pragma unroll
    for (int c = 0; c < 2; ++c) {
      const _Float16* pr = &sP[wave][m * 64 + c * 32];
      pf[c] = cat8(*(const v8h*)(pr + kb), *(const v8h*)(pr + kb + 8));
    }

    // O += P @ V : B element [k=key, n=e] = sVt[e][key], contiguous per lane.
#pragma unroll
    for (int t = 0; t < 4; ++t) {
#pragma unroll
      for (int c = 0; c < 2; ++c) {
        const v16h bv = *(const v16h*)(&sVt[cur][0] +
                                       (size_t)(t * 16 + m) * 64 + c * 32 +
                                       hl * 16);
        o[t] = wmma_f16(pf[c], bv, o[t]);
      }
    }
    __syncthreads();  // all reads of buf[cur] done before it is re-filled
  }

  // Epilogue: normalize, nontemporal fp32 store (write-once output).
#pragma unroll
  for (int r = 0; r < 8; ++r) {
    const float inv = 1.0f / lrow[r];
    float* orow = Out + ((size_t)bIdx * SEQ + qRowBase + mo + r) * D_OUT;
#pragma unroll
    for (int t = 0; t < 4; ++t)
      __builtin_nontemporal_store(o[t][r] * inv, orow + t * 16 + m);
  }
}

// ---------------------------------------------------------------------------
extern "C" void kernel_launch(void* const* d_in, const int* in_sizes, int n_in,
                              void* d_out, int out_size, void* d_ws,
                              size_t ws_size, hipStream_t stream) {
  const float* Xk = (const float*)d_in[0];
  const float* Xv = (const float*)d_in[1];
  const float* Xq = (const float*)d_in[2];
  const float* WK = (const float*)d_in[3];
  const float* WV = (const float*)d_in[4];
  const float* WQ = (const float*)d_in[5];

  const size_t tokens = (size_t)BATCH * SEQ;  // 16384
  _Float16* Kh = (_Float16*)d_ws;             // [B*S][64]   2 MB
  _Float16* Vt = Kh + tokens * D_OUT;         // [B*64][S]   2 MB
  _Float16* Qh = Vt + tokens * D_OUT;         // [B*S][64]   2 MB

  dim3 pgrid((unsigned)(tokens / 128));
  proj_f16_kernel<<<pgrid, 256, 0, stream>>>(Xk, WK, Kh, 0);
  proj_f16_kernel<<<pgrid, 256, 0, stream>>>(Xv, WV, Vt, 1);
  proj_f16_kernel<<<pgrid, 256, 0, stream>>>(Xq, WQ, Qh, 0);

  dim3 agrid(SEQ / 128, BATCH);
  flash_attn_kernel<<<agrid, 256, 0, stream>>>(Qh, Kh, Vt, (float*)d_out);
}
